// FeatureDGCNN_52553219833918
// MI455X (gfx1250) — matrix operand order repaired and checked by reading, not compiled, not run on hardware
//
#include <hip/hip_runtime.h>
#include <hip/hip_bf16.h>

#define BB 8
#define NN 2048
#define KNBR 20
#define SLOPEV 0.2f
#define EPSV 1e-5f

typedef __attribute__((ext_vector_type(16))) __bf16 v16bf;
typedef __attribute__((ext_vector_type(8)))  float  v8f;
typedef __attribute__((ext_vector_type(4)))  unsigned int u32x4;
typedef __attribute__((ext_vector_type(8)))  int  i32x8;
typedef __attribute__((ext_vector_type(4)))  int  i32x4;

union Frag16 { uint4 q[2]; unsigned short s[16]; v16bf v; };

__device__ __forceinline__ unsigned short f2bf(float x) {
  unsigned int u = __float_as_uint(x);
  u += 0x7fffu + ((u >> 16) & 1u);   // round-to-nearest-even
  return (unsigned short)(u >> 16);
}
__device__ __forceinline__ float lrelu(float y) { return y >= 0.f ? y : SLOPEV * y; }

// order-preserving f32 -> u32 bijection (for ds_max_u32 float max)
__device__ __forceinline__ unsigned enc_ord(float x) {
  unsigned u = __float_as_uint(x);
  return ((int)u < 0) ? ~u : (u | 0x80000000u);
}
__device__ __forceinline__ float dec_ord(unsigned e) {
  unsigned u = (e & 0x80000000u) ? (e ^ 0x80000000u) : ~e;
  return __uint_as_float(u);
}

// ---------------- prep kernels ----------------
__global__ void cvt_bf16_kernel(unsigned short* __restrict__ dst,
                                const float* __restrict__ src, int n) {
  int i = blockIdx.x * blockDim.x + threadIdx.x;
  if (i < n) dst[i] = f2bf(src[i]);
}

__global__ void cvt_bf16_pad_kernel(unsigned short* __restrict__ dst,
                                    const float* __restrict__ src,
                                    int O, int Cin, int Cpad) {
  int i = blockIdx.x * blockDim.x + threadIdx.x;
  if (i < O * Cpad) {
    int o = i / Cpad, c = i - o * Cpad;
    dst[i] = (c < Cin) ? f2bf(src[o * Cin + c]) : (unsigned short)0;
  }
}

__global__ void bnprep_kernel(float* __restrict__ scale, float* __restrict__ shift,
                              const float* __restrict__ g, const float* __restrict__ b,
                              const float* __restrict__ m, const float* __restrict__ v,
                              int n) {
  int i = blockIdx.x * blockDim.x + threadIdx.x;
  if (i < n) {
    float s = g[i] * rsqrtf(v[i] + EPSV);
    scale[i] = s;
    shift[i] = b[i] - m[i] * s;
  }
}

__global__ void transpose_x_kernel(float* __restrict__ xt, const float* __restrict__ x) {
  int i = blockIdx.x * blockDim.x + threadIdx.x;   // B*3*N threads
  if (i < BB * 3 * NN) {
    int b = i / (3 * NN);
    int r = i - b * 3 * NN;
    int c = r / NN;
    int n = r - c * NN;
    xt[((size_t)b * NN + n) * 3 + c] = x[i];
  }
}

// ---------------- KNN ----------------
__global__ void knn_kernel(const float* __restrict__ xt, int* __restrict__ idx) {
  __shared__ float pts[NN * 3];
  const int b = blockIdx.y;
  const int n = blockIdx.x * blockDim.x + threadIdx.x;
  for (int i = threadIdx.x; i < NN * 3; i += blockDim.x)
    pts[i] = xt[(size_t)b * NN * 3 + i];
  __syncthreads();
  const float qx = pts[n * 3 + 0], qy = pts[n * 3 + 1], qz = pts[n * 3 + 2];
  float bd[KNBR]; int bix[KNBR];
#pragma unroll
  for (int k = 0; k < KNBR; ++k) { bd[k] = -3.4e38f; bix[k] = 0; }
  for (int j = 0; j < NN; ++j) {
    float dx = pts[j * 3 + 0] - qx, dy = pts[j * 3 + 1] - qy, dz = pts[j * 3 + 2] - qz;
    float nd = -(dx * dx + dy * dy + dz * dz);   // larger = nearer
    if (nd > bd[KNBR - 1]) {
      float cd = nd; int cidx = j;
#pragma unroll
      for (int t = 0; t < KNBR; ++t) {           // register insertion (stable)
        bool sw = cd > bd[t];
        float td = sw ? bd[t] : cd; int ti = sw ? bix[t] : cidx;
        bd[t] = sw ? cd : bd[t];    bix[t] = sw ? cidx : bix[t];
        cd = td; cidx = ti;
      }
    }
  }
#pragma unroll
  for (int k = 0; k < KNBR; ++k)
    idx[((size_t)b * NN + n) * KNBR + k] = bix[k];
}

// ---------------- EdgeConv v2: LDS-staged edge matrix + WMMA + ds-atomic max
template <int CI, int CPAD, int COUT>
__global__ void edgeconv2_kernel(const float* __restrict__ hin, int cinStride, int choff,
                                 const unsigned short* __restrict__ Wb,
                                 const float* __restrict__ bnscale,
                                 const float* __restrict__ bnshift,
                                 const int* __restrict__ idx,
                                 float* __restrict__ hout_f32,
                                 unsigned short* __restrict__ hout_bf,
                                 int outoff) {
  __shared__ unsigned short E[80 * CPAD];     // edge matrix chunk, bf16
  __shared__ unsigned int   mbuf[COUT * 16];  // ordered-u32 running max (o, p)
  const int tid  = threadIdx.x;
  const int lane = tid & 31;
  const int wv   = tid >> 5;
  const int b    = blockIdx.z;
  const int ntile= blockIdx.x;
  const int kbA  = (lane >> 4) * 16;          // A: contiguous k = 16*(lane/16) + s
  const int kbB  = (lane >> 4) * 16;          // B: same contiguous k order as A

  for (int i = tid; i < COUT * 16; i += 256) mbuf[i] = 0u;

  for (int chunk = 0; chunk < 4; ++chunk) {
    __syncthreads();                          // protect E from previous chunk
    constexpr int G = CPAD / 8;
    for (int g = tid; g < 80 * G; g += 256) {
      int colc = g % 80;
      int cg   = (g / 80) * 8;
      int col  = chunk * 80 + colc;
      int p = col / 20, kk = col - p * 20;
      int n = ntile * 16 + p;
      int nbr = idx[((size_t)b * NN + n) * KNBR + kk];
      const float* rowc = hin + ((size_t)b * NN + n)   * cinStride + choff;
      const float* rown = hin + ((size_t)b * NN + nbr) * cinStride + choff;
      union { unsigned short s[8]; uint4 q; } pk;
      if constexpr (CI < 8) {                 // layer 1: tiny, scalar path
#pragma unroll
        for (int j = 0; j < 8; ++j) {
          int c = cg + j;
          float v = (c < CI) ? rown[c] - rowc[c]
                             : ((c < 2 * CI) ? rowc[c - CI] : 0.f);
          pk.s[j] = f2bf(v);
        }
      } else {                                // CI % 8 == 0: region uniform per group
        float4 a0, a1;
        if (cg < CI) {
          a0 = *(const float4*)(rown + cg);
          a1 = *(const float4*)(rown + cg + 4);
          float4 c0 = *(const float4*)(rowc + cg);
          float4 c1 = *(const float4*)(rowc + cg + 4);
          a0.x -= c0.x; a0.y -= c0.y; a0.z -= c0.z; a0.w -= c0.w;
          a1.x -= c1.x; a1.y -= c1.y; a1.z -= c1.z; a1.w -= c1.w;
        } else {
          a0 = *(const float4*)(rowc + cg - CI);
          a1 = *(const float4*)(rowc + cg - CI + 4);
        }
        pk.s[0] = f2bf(a0.x); pk.s[1] = f2bf(a0.y);
        pk.s[2] = f2bf(a0.z); pk.s[3] = f2bf(a0.w);
        pk.s[4] = f2bf(a1.x); pk.s[5] = f2bf(a1.y);
        pk.s[6] = f2bf(a1.z); pk.s[7] = f2bf(a1.w);
      }
      *(uint4*)(E + colc * CPAD + cg) = pk.q;  // ds_store_b128
    }
    __syncthreads();
    const int tiles = 5 * (COUT / 16);
    for (int t = wv; t < tiles; t += 8) {
      const int ct    = t % 5;
      const int otile = (t / 5) * 16;
      const int colc  = ct * 16 + (lane & 15);
      const int p     = (chunk * 80 + colc) / 20;
      const unsigned short* wrow = Wb + (size_t)(otile + (lane & 15)) * CPAD;
      const unsigned short* erow = E + colc * CPAD;
      v8f acc = {};
#pragma unroll
      for (int ks = 0; ks < CPAD; ks += 32) {
        Frag16 fa, fb;
        fa.q[0] = *(const uint4*)(wrow + ks + kbA);       // global_load_b128
        fa.q[1] = *(const uint4*)(wrow + ks + kbA + 8);
        fb.q[0] = *(const uint4*)(erow + ks + kbB);       // ds_load_b128
        fb.q[1] = *(const uint4*)(erow + ks + kbB + 8);
        acc = __builtin_amdgcn_wmma_f32_16x16x32_bf16(false, fa.v, false, fb.v,
                                                      (short)0, acc, false, false);
      }
      const int mloc = (lane >> 4) * 8;
#pragma unroll
      for (int r = 0; r < 8; ++r) {
        int o = otile + mloc + r;
        float y = lrelu(acc[r] * bnscale[o] + bnshift[o]);
        atomicMax(&mbuf[o * 16 + p], enc_ord(y));         // ds_max_u32
      }
    }
  }
  __syncthreads();
  for (int cell = tid; cell < COUT * 16; cell += 256) {
    int o = cell >> 4, p = cell & 15;
    float mx = dec_ord(mbuf[cell]);
    size_t off = ((size_t)b * NN + ntile * 16 + p) * 512 + outoff + o;
    hout_f32[off] = mx;
    hout_bf[off]  = f2bf(mx);
  }
}

// ---------------- Pointwise WMMA GEMM: TDM weight staging + pipelined B ----
// Block = (16 out-ch tile, 64 columns, batch), 128 threads (4 waves).
// Wave 0 DMAs the 16 x Cin bf16 weight tile to LDS via tensor_load_to_lds,
// waits TENSORcnt, barrier; all waves then read A-fragments with ds_load_b128
// while B-fragments stream from global with a 1-stage register pipeline.
__global__ void pw_gemm_kernel(const unsigned short* __restrict__ Hin, int Cin,
                               const unsigned short* __restrict__ Wb,
                               const float* __restrict__ bias,
                               const float* __restrict__ bnscale,
                               const float* __restrict__ bnshift,
                               unsigned short* __restrict__ out_bf,
                               float* __restrict__ out_cf,
                               int Cout) {
  __shared__ unsigned short Wlds[16 * 1024];   // up to 32 KB weight tile
  const int lane  = threadIdx.x & 31;
  const int wv    = threadIdx.x >> 5;
  const int b     = blockIdx.z;
  const int col   = blockIdx.x * 64 + wv * 16 + (lane & 15);
  const int otile = blockIdx.y * 16;
  const int kbA   = (lane >> 4) * 16;
  const int kbB   = (lane >> 4) * 16;

  if (wv == 0) {                               // one TDM per block (per-wave cnt)
    unsigned nd  = (unsigned)(16 * Cin * 2) >> 2;          // payload in dwords
    unsigned long long ga = (unsigned long long)(uintptr_t)(Wb + (size_t)otile * Cin);
    unsigned lds = (unsigned)(uintptr_t)(void*)Wlds;       // LDS byte offset
    u32x4 g0; i32x8 g1;
    i32x4 g2 = {0, 0, 0, 0}, g3 = {0, 0, 0, 0};
    i32x8 g4 = {0, 0, 0, 0, 0, 0, 0, 0};
    g0[0] = 1u;                                            // D# count=1, user mode
    g0[1] = lds;                                           // lds_addr
    g0[2] = (unsigned)ga;                                  // global_addr[31:0]
    g0[3] = (unsigned)((ga >> 32) & 0x1FFFFFFu) | (2u << 30); // addr[56:32] | type=2
    g1[0] = 0x20000;                                       // data_size = 4B dwords
    g1[1] = (int)((nd & 0xFFFFu) << 16);                   // tensor_dim0[15:0]
    g1[2] = (int)((nd >> 16) | 0x10000u);                  // td0[31:16] | td1=1
    g1[3] = (int)(nd << 16);                               // tile_dim0 (1-D copy)
    g1[4] = 0;
    g1[5] = (int)nd;                                       // tensor_dim0_stride
    g1[6] = 0; g1[7] = 0;
    __builtin_amdgcn_tensor_load_to_lds(g0, g1, g2, g3, g4, 0);
    __builtin_amdgcn_s_wait_tensorcnt(0);
  }
  __syncthreads();

  const unsigned short* wrow = Wlds + (lane & 15) * Cin;   // LDS A rows
  const unsigned short* hrow = Hin + ((size_t)b * NN + col) * Cin;

  v8f acc = {};
  Frag16 fb0;
  fb0.q[0] = *(const uint4*)(hrow + kbB);
  fb0.q[1] = *(const uint4*)(hrow + kbB + 8);
#pragma unroll 4
  for (int ks = 0; ks < Cin; ks += 32) {
    Frag16 fb1;                                // prefetch next B (reg pipeline)
    fb1.q[0] = *(const uint4*)(hrow + ks + 32 + kbB);
    fb1.q[1] = *(const uint4*)(hrow + ks + 32 + kbB + 8);
    __builtin_prefetch(hrow + ks + 128, 0, 1);
    Frag16 fa;                                 // A from LDS (ds_load_b128)
    fa.q[0] = *(const uint4*)(wrow + ks + kbA);
    fa.q[1] = *(const uint4*)(wrow + ks + kbA + 8);
    acc = __builtin_amdgcn_wmma_f32_16x16x32_bf16(false, fa.v, false, fb0.v,
                                                  (short)0, acc, false, false);
    fb0 = fb1;
  }
  const int mloc = (lane >> 4) * 8;
  const int ob   = otile + mloc;
  float yv[8];
#pragma unroll
  for (int r = 0; r < 8; ++r) yv[r] = acc[r];
  if (bias) {
#pragma unroll
    for (int r = 0; r < 8; ++r) yv[r] += bias[ob + r];
  }
  if (bnscale) {
#pragma unroll
    for (int r = 0; r < 8; ++r) yv[r] = lrelu(yv[r] * bnscale[ob + r] + bnshift[ob + r]);
  }
  if (out_bf) {
    union { unsigned short s[8]; uint4 q; } pk;
#pragma unroll
    for (int r = 0; r < 8; ++r) pk.s[r] = f2bf(yv[r]);
    *(uint4*)(out_bf + ((size_t)b * NN + col) * Cout + ob) = pk.q;
  }
  if (out_cf) {
#pragma unroll
    for (int r = 0; r < 8; ++r)
      out_cf[((size_t)b * Cout + ob + r) * NN + col] = yv[r];
  }
}

// ---------------------------------------------------------------------------
extern "C" void kernel_launch(void* const* d_in, const int* in_sizes, int n_in,
                              void* d_out, int out_size, void* d_ws, size_t ws_size,
                              hipStream_t stream) {
  (void)in_sizes; (void)n_in; (void)out_size; (void)ws_size;
  const float* x = (const float*)d_in[0];
  const float* encW[4]  = {(const float*)d_in[1], (const float*)d_in[2],
                           (const float*)d_in[3], (const float*)d_in[4]};
  const float* encBN[4][4];
  for (int i = 0; i < 4; ++i)
    for (int j = 0; j < 4; ++j) encBN[i][j] = (const float*)d_in[5 + 4 * i + j];
  const float* lastW = (const float*)d_in[21];
  const float* lastBN[4] = {(const float*)d_in[22], (const float*)d_in[23],
                            (const float*)d_in[24], (const float*)d_in[25]};
  const float* featW[3] = {(const float*)d_in[26], (const float*)d_in[27],
                           (const float*)d_in[28]};
  const float* featB[3] = {(const float*)d_in[29], (const float*)d_in[30],
                           (const float*)d_in[31]};
  const float* featBN[2][4];
  for (int i = 0; i < 2; ++i)
    for (int j = 0; j < 4; ++j) featBN[i][j] = (const float*)d_in[32 + 4 * i + j];

  char* base = (char*)d_ws;
  size_t off = 0;
  auto alloc = [&](size_t bytes) {
    void* p = base + off;
    off = (off + bytes + 255) & ~(size_t)255;
    return p;
  };
  int*            idxb   = (int*)   alloc((size_t)BB * NN * KNBR * 4);
  float*          xt     = (float*) alloc((size_t)BB * NN * 3 * 4);
  float*          cat32  = (float*) alloc((size_t)BB * NN * 512 * 4);
  unsigned short* catbf  = (unsigned short*)alloc((size_t)BB * NN * 512 * 2);
  unsigned short* act1   = (unsigned short*)alloc((size_t)BB * NN * 1024 * 2);
  unsigned short* act2   = (unsigned short*)alloc((size_t)BB * NN * 512 * 2);
  unsigned short* act3   = (unsigned short*)alloc((size_t)BB * NN * 128 * 2);
  unsigned short* wbE0   = (unsigned short*)alloc(64 * 32 * 2);
  unsigned short* wbE1   = (unsigned short*)alloc(64 * 128 * 2);
  unsigned short* wbE2   = (unsigned short*)alloc(128 * 128 * 2);
  unsigned short* wbE3   = (unsigned short*)alloc(256 * 256 * 2);
  unsigned short* wbL    = (unsigned short*)alloc(1024 * 512 * 2);
  unsigned short* wbF0   = (unsigned short*)alloc(512 * 1024 * 2);
  unsigned short* wbF1   = (unsigned short*)alloc(128 * 512 * 2);
  unsigned short* wbF2   = (unsigned short*)alloc(32 * 128 * 2);
  float* sE[4]; float* tE[4];
  int encC[4] = {64, 64, 128, 256};
  for (int i = 0; i < 4; ++i) { sE[i] = (float*)alloc(encC[i] * 4); tE[i] = (float*)alloc(encC[i] * 4); }
  float* sL  = (float*)alloc(1024 * 4); float* tL  = (float*)alloc(1024 * 4);
  float* sF0 = (float*)alloc(512 * 4);  float* tF0 = (float*)alloc(512 * 4);
  float* sF1 = (float*)alloc(128 * 4);  float* tF1 = (float*)alloc(128 * 4);

  // ---- prep ----
  transpose_x_kernel<<<(BB * 3 * NN + 255) / 256, 256, 0, stream>>>(xt, x);
  cvt_bf16_pad_kernel<<<(64 * 32 + 255) / 256, 256, 0, stream>>>(wbE0, encW[0], 64, 6, 32);
  cvt_bf16_kernel<<<(64 * 128 + 255) / 256, 256, 0, stream>>>(wbE1, encW[1], 64 * 128);
  cvt_bf16_kernel<<<(128 * 128 + 255) / 256, 256, 0, stream>>>(wbE2, encW[2], 128 * 128);
  cvt_bf16_kernel<<<(256 * 256 + 255) / 256, 256, 0, stream>>>(wbE3, encW[3], 256 * 256);
  cvt_bf16_kernel<<<(1024 * 512 + 255) / 256, 256, 0, stream>>>(wbL, lastW, 1024 * 512);
  cvt_bf16_kernel<<<(512 * 1024 + 255) / 256, 256, 0, stream>>>(wbF0, featW[0], 512 * 1024);
  cvt_bf16_kernel<<<(128 * 512 + 255) / 256, 256, 0, stream>>>(wbF1, featW[1], 128 * 512);
  cvt_bf16_kernel<<<(32 * 128 + 255) / 256, 256, 0, stream>>>(wbF2, featW[2], 32 * 128);
  for (int i = 0; i < 4; ++i)
    bnprep_kernel<<<(encC[i] + 255) / 256, 256, 0, stream>>>(
        sE[i], tE[i], encBN[i][0], encBN[i][1], encBN[i][2], encBN[i][3], encC[i]);
  bnprep_kernel<<<4, 256, 0, stream>>>(sL, tL, lastBN[0], lastBN[1], lastBN[2], lastBN[3], 1024);
  bnprep_kernel<<<2, 256, 0, stream>>>(sF0, tF0, featBN[0][0], featBN[0][1], featBN[0][2], featBN[0][3], 512);
  bnprep_kernel<<<1, 256, 0, stream>>>(sF1, tF1, featBN[1][0], featBN[1][1], featBN[1][2], featBN[1][3], 128);

  // ---- KNN ----
  knn_kernel<<<dim3(NN / 256, BB), 256, 0, stream>>>(xt, idxb);

  // ---- EdgeConv layers (LDS-staged WMMA) ----
  edgeconv2_kernel<3, 32, 64><<<dim3(NN / 16, 1, BB), 256, 0, stream>>>(
      xt, 3, 0, wbE0, sE[0], tE[0], idxb, cat32, catbf, 0);
  edgeconv2_kernel<64, 128, 64><<<dim3(NN / 16, 1, BB), 256, 0, stream>>>(
      cat32, 512, 0, wbE1, sE[1], tE[1], idxb, cat32, catbf, 64);
  edgeconv2_kernel<64, 128, 128><<<dim3(NN / 16, 1, BB), 256, 0, stream>>>(
      cat32, 512, 64, wbE2, sE[2], tE[2], idxb, cat32, catbf, 128);
  edgeconv2_kernel<128, 256, 256><<<dim3(NN / 16, 1, BB), 256, 0, stream>>>(
      cat32, 512, 128, wbE3, sE[3], tE[3], idxb, cat32, catbf, 256);

  // ---- Pointwise chain (TDM + WMMA) ----
  pw_gemm_kernel<<<dim3(NN / 64, 1024 / 16, BB), 128, 0, stream>>>(
      catbf, 512, wbL, nullptr, sL, tL, act1, nullptr, 1024);
  pw_gemm_kernel<<<dim3(NN / 64, 512 / 16, BB), 128, 0, stream>>>(
      act1, 1024, wbF0, featB[0], sF0, tF0, act2, nullptr, 512);
  pw_gemm_kernel<<<dim3(NN / 64, 128 / 16, BB), 128, 0, stream>>>(
      act2, 512, wbF1, featB[1], sF1, tF1, act3, nullptr, 128);
  pw_gemm_kernel<<<dim3(NN / 64, 32 / 16, BB), 128, 0, stream>>>(
      act3, 128, wbF2, featB[2], nullptr, nullptr, nullptr, (float*)d_out, 32);
}
